// HolonomyAttention_7533372637367
// MI455X (gfx1250) — matrix operand order repaired and checked by reading, not compiled, not run on hardware
//
#include <hip/hip_runtime.h>

// ---------------------------------------------------------------------------
// HolonomyAttention for MI455X (gfx1250): per-head rotation + causal flash
// attention. All GEMMs on v_wmma_f32_16x16x32_bf16; fragments fed with
// ds_load_b128 (K-contiguous LDS tiling + padded strides); fp32 softmax.
// TKV=64 amortizes softmax bookkeeping over twice the columns per barrier.
// ---------------------------------------------------------------------------

typedef __attribute__((ext_vector_type(16))) __bf16          v16bf;
typedef __attribute__((ext_vector_type(16))) unsigned short  v16u;
typedef __attribute__((ext_vector_type(8)))  unsigned short  v8u;
typedef __attribute__((ext_vector_type(8)))  float           v8f;

#define BATCH 2
#define HEADS 16
#define SEQ   2048
#define DHEAD 64
#define TQ    128   // q rows per workgroup (8 waves x 16 rows)
#define TKV   64    // kv tile width (2x bf16 WMMA K depth)
#define NW    8     // waves per workgroup

// padded LDS row stride (ushorts): 144 B/row keeps 16B alignment for b128
// loads while spreading lanes across bank groups
#define LDQ 72

__device__ __forceinline__ unsigned short f32_to_bf16(float f) {
  unsigned int u = __builtin_bit_cast(unsigned int, f);
  u = (u + 0x7FFFu + ((u >> 16) & 1u)) >> 16;   // round-to-nearest-even
  return (unsigned short)u;
}

__device__ __forceinline__ unsigned int pack2_bf16(float a, float b) {
  return (unsigned int)f32_to_bf16(a) | ((unsigned int)f32_to_bf16(b) << 16);
}

__device__ __forceinline__ v16bf cat_bf(v8u lo, v8u hi) {
  union { struct { v8u lo, hi; } h; v16u u; v16bf b; } c;
  c.h.lo = lo; c.h.hi = hi;
  return c.b;
}

// A-matrix fragment (16x32 bf16), tile row-major with K contiguous.
// Lane L: M=L%16, half=L/16. Two 16B ds_load_b128 per lane.
__device__ __forceinline__ v16bf load_A(const unsigned short* lds, int r0,
                                        int ld, int k0, int lane) {
  const int m = lane & 15, half = lane >> 4;
  const unsigned short* p = lds + (r0 + m) * ld + k0 + half * 8;
  const v8u lo = *(const v8u*)p;
  const v8u hi = *(const v8u*)(p + 16);
  return cat_bf(lo, hi);
}

// B-matrix fragment (32x16 bf16) from a tile stored [N][K] (K contiguous).
// Lane L: N=L%16, half=L/16 -> K in [kb+8h, kb+8h+7] and [kb+16+8h, kb+16+8h+7]: two b128 loads.
__device__ __forceinline__ v16bf load_B(const unsigned short* lds, int nb,
                                        int kb, int ld, int lane) {
  const int n = lane & 15, half = lane >> 4;
  const unsigned short* p = lds + (nb + n) * ld + kb + half * 8;
  const v8u lo = *(const v8u*)p;
  const v8u hi = *(const v8u*)(p + 16);
  return cat_bf(lo, hi);
}

__device__ __forceinline__ v8f wmma_bf16(v16bf a, v16bf b, v8f c) {
  return __builtin_amdgcn_wmma_f32_16x16x32_bf16(
      false, a, false, b, (short)0, c, false, false);
}

// Stage one K/V tile in 2x2 blocks: float2 global loads, packed b32 LDS
// stores. K as [s][d] (K-dim d contiguous), V transposed [d][s] (K-dim s
// contiguous).
__device__ __forceinline__ void stage_kv(unsigned short* dK, unsigned short* dVt,
                                         const float* Kp, const float* Vp,
                                         int s0, int tid) {
#pragma unroll
  for (int it = 0; it < (TKV * DHEAD / 4) / 256; ++it) {
    const int bi = tid + 256 * it;       // 2x2 block index
    const int s = (bi >> 5) * 2;         // 0..62
    const int d = (bi & 31) * 2;         // 0..62
    const float2 ka = *(const float2*)(Kp + (size_t)(s0 + s) * DHEAD + d);
    const float2 kb = *(const float2*)(Kp + (size_t)(s0 + s + 1) * DHEAD + d);
    const float2 va = *(const float2*)(Vp + (size_t)(s0 + s) * DHEAD + d);
    const float2 vb = *(const float2*)(Vp + (size_t)(s0 + s + 1) * DHEAD + d);
    *(unsigned int*)(dK + s * LDQ + d)        = pack2_bf16(ka.x, ka.y);
    *(unsigned int*)(dK + (s + 1) * LDQ + d)  = pack2_bf16(kb.x, kb.y);
    *(unsigned int*)(dVt + d * LDQ + s)       = pack2_bf16(va.x, vb.x);
    *(unsigned int*)(dVt + (d + 1) * LDQ + s) = pack2_bf16(va.y, vb.y);
  }
}

__launch_bounds__(256, 1)
__global__ void holonomy_attn_kernel(const float* __restrict__ Q,
                                     const float* __restrict__ K,
                                     const float* __restrict__ V,
                                     const float* __restrict__ Curv,
                                     float* __restrict__ Out) {
  __shared__ __attribute__((aligned(16))) unsigned short sQ [TQ * LDQ];       // 18 KB
  __shared__ __attribute__((aligned(16))) unsigned short sCt[DHEAD * LDQ];    //  9 KB
  __shared__ __attribute__((aligned(16))) unsigned short sQr[TQ * LDQ];       // 18 KB
  __shared__ __attribute__((aligned(16))) unsigned short sK [2][TKV * LDQ];   // 18 KB
  __shared__ __attribute__((aligned(16))) unsigned short sVt[2][DHEAD * LDQ]; // 18 KB
  __shared__ __attribute__((aligned(16))) unsigned short sP [NW * 16 * LDQ];  // 18 KB

  const int tid  = threadIdx.x;
  const int lane = tid & 31;
  const int w    = tid >> 5;
  const int m    = lane & 15;
  const int half = lane >> 4;

  const int qb = blockIdx.x * TQ;
  const int h  = blockIdx.y;
  const int b  = blockIdx.z;

  const size_t bh = ((size_t)b * HEADS + h) * SEQ;
  const float* Qp = Q + (bh + qb) * DHEAD;
  const float* Kp = K + bh * DHEAD;
  const float* Vp = V + bh * DHEAD;
  const float* Cp = Curv + (size_t)h * DHEAD * DHEAD;
  float*       Op = Out + (bh + qb) * DHEAD;

  // ---- stage Q (row-major, packed b32 stores) and C^T as bf16 ------------
  for (int i = tid; i < TQ * DHEAD / 2; i += 256) {
    const int r = i >> 5, c = (i & 31) * 2;
    const float2 qv = *(const float2*)(Qp + (size_t)r * DHEAD + c);
    *(unsigned int*)(sQ + r * LDQ + c) = pack2_bf16(qv.x, qv.y);
  }
  for (int i = tid; i < DHEAD * DHEAD; i += 256) {
    const int d = i >> 6, e = i & 63;          // C[d][e] -> Ct[e][d]
    sCt[e * LDQ + d] = f32_to_bf16(Cp[i]);
  }
  __syncthreads();

  // ---- Q_rot = (Q @ C) * scale ------------------------------------------
  const float scale = 0.125f;  // 1/sqrt(64)
  const int r0 = w * 16;
  {
    const v16bf aq0 = load_A(sQ, r0, LDQ, 0, lane);
    const v16bf aq1 = load_A(sQ, r0, LDQ, 32, lane);
#pragma unroll
    for (int j = 0; j < 4; ++j) {
      v8f acc = {};
      acc = wmma_bf16(aq0, load_B(sCt, j * 16, 0,  LDQ, lane), acc);
      acc = wmma_bf16(aq1, load_B(sCt, j * 16, 32, LDQ, lane), acc);
#pragma unroll
      for (int r = 0; r < 8; ++r) {
        const int row = r0 + r + 8 * half;     // C/D layout
        sQr[row * LDQ + j * 16 + m] = f32_to_bf16(acc[r] * scale);
      }
    }
  }
  __syncthreads();

  const v16bf a0 = load_A(sQr, r0, LDQ, 0, lane);
  const v16bf a1 = load_A(sQr, r0, LDQ, 32, lane);

  v8f o[4] = {v8f{}, v8f{}, v8f{}, v8f{}};
  float mrow[8], lrow[8];
#pragma unroll
  for (int r = 0; r < 8; ++r) { mrow[r] = -1e30f; lrow[r] = 0.0f; }

  unsigned short* myP = sP + w * 16 * LDQ;
  const int nt = (qb + TQ) / TKV;   // causal: kv tiles fully <= q block end

  stage_kv(sK[0], sVt[0], Kp, Vp, 0, tid);
  __syncthreads();

  for (int t = 0; t < nt; ++t) {
    const int cur = t & 1;
    const int s0  = t * TKV;

    // stage next tile into the other buffer (overlaps with compute)
    if (t + 1 < nt) {
      stage_kv(sK[cur ^ 1], sVt[cur ^ 1], Kp, Vp, s0 + TKV, tid);
      if (t + 2 < nt) {  // prefetch tile t+2 -> global_prefetch_b8
        __builtin_prefetch(Kp + (size_t)(s0 + 2 * TKV) * DHEAD + tid * 16, 0, 0);
        __builtin_prefetch(Vp + (size_t)(s0 + 2 * TKV) * DHEAD + tid * 16, 0, 0);
      }
    }

    // ---- scores S = Qrot @ K^T: four 16x16 subtiles over s ---------------
    v8f s[4];
#pragma unroll
    for (int jj = 0; jj < 4; ++jj) {
      v8f acc = {};
      acc = wmma_bf16(a0, load_B(sK[cur], jj * 16, 0,  LDQ, lane), acc);
      acc = wmma_bf16(a1, load_B(sK[cur], jj * 16, 32, LDQ, lane), acc);
      s[jj] = acc;
    }

    // ---- causal mask + online softmax over 64 columns --------------------
#pragma unroll
    for (int r = 0; r < 8; ++r) {
      const int q = qb + r0 + r + 8 * half;
      const float v0 = (s0 + m      <= q) ? s[0][r] : -1e30f;
      const float v1 = (s0 + 16 + m <= q) ? s[1][r] : -1e30f;
      const float v2 = (s0 + 32 + m <= q) ? s[2][r] : -1e30f;
      const float v3 = (s0 + 48 + m <= q) ? s[3][r] : -1e30f;

      float tmax = fmaxf(fmaxf(v0, v1), fmaxf(v2, v3));
      tmax = fmaxf(tmax, __shfl_xor(tmax, 1));
      tmax = fmaxf(tmax, __shfl_xor(tmax, 2));
      tmax = fmaxf(tmax, __shfl_xor(tmax, 4));
      tmax = fmaxf(tmax, __shfl_xor(tmax, 8));

      const float mnew  = fmaxf(mrow[r], tmax);
      const float alpha = __expf(mrow[r] - mnew);
      const float p0 = __expf(v0 - mnew);
      const float p1 = __expf(v1 - mnew);
      const float p2 = __expf(v2 - mnew);
      const float p3 = __expf(v3 - mnew);

      float psum = (p0 + p1) + (p2 + p3);
      psum += __shfl_xor(psum, 1);
      psum += __shfl_xor(psum, 2);
      psum += __shfl_xor(psum, 4);
      psum += __shfl_xor(psum, 8);

      lrow[r] = lrow[r] * alpha + psum;
      mrow[r] = mnew;
#pragma unroll
      for (int j = 0; j < 4; ++j) o[j][r] *= alpha;

      const int prow = r + 8 * half;           // restage P (C/D -> A layout)
      myP[prow * LDQ + m]      = f32_to_bf16(p0);
      myP[prow * LDQ + 16 + m] = f32_to_bf16(p1);
      myP[prow * LDQ + 32 + m] = f32_to_bf16(p2);
      myP[prow * LDQ + 48 + m] = f32_to_bf16(p3);
    }

    // ---- O += P @ V (K-dim s = 64: two chunks) ---------------------------
    const v16bf ap0 = load_A(myP, 0, LDQ, 0, lane);
    const v16bf ap1 = load_A(myP, 0, LDQ, 32, lane);
#pragma unroll
    for (int j = 0; j < 4; ++j) {
      o[j] = wmma_bf16(ap0, load_B(sVt[cur], j * 16, 0,  LDQ, lane), o[j]);
      o[j] = wmma_bf16(ap1, load_B(sVt[cur], j * 16, 32, LDQ, lane), o[j]);
    }

    __syncthreads();   // all waves done with cur; next buffer fully staged
  }

  // ---- normalize and write out (fp32) ------------------------------------
#pragma unroll
  for (int r = 0; r < 8; ++r) {
    const float inv = 1.0f / lrow[r];
    const int row = r0 + r + 8 * half;
#pragma unroll
    for (int j = 0; j < 4; ++j)
      Op[(size_t)row * DHEAD + j * 16 + m] = o[j][r] * inv;
  }
}

extern "C" void kernel_launch(void* const* d_in, const int* in_sizes, int n_in,
                              void* d_out, int out_size, void* d_ws, size_t ws_size,
                              hipStream_t stream) {
  (void)in_sizes; (void)n_in; (void)out_size; (void)d_ws; (void)ws_size;
  const float* Q    = (const float*)d_in[0];
  const float* K    = (const float*)d_in[1];
  const float* V    = (const float*)d_in[2];
  // d_in[3] is the causal mask; computed analytically in-kernel.
  const float* Curv = (const float*)d_in[4];
  float* Out = (float*)d_out;

  dim3 grid(SEQ / TQ, HEADS, BATCH);   // (16, 16, 2)
  dim3 block(256);                     // 8 waves (wave32)
  holonomy_attn_kernel<<<grid, block, 0, stream>>>(Q, K, V, Curv, Out);
}
